// Pro_jts_02_pccp_58720792871473
// MI455X (gfx1250) — compile-verified, matrix operand order checked by reading, not run on hardware
//
#include <hip/hip_runtime.h>

typedef __attribute__((ext_vector_type(16))) _Float16 v16h;
typedef __attribute__((ext_vector_type(8)))  float    v8f;

#define NBONE 16
#define FEPS 1e-9f

static constexpr int kPar[NBONE] = {0,1,2,0,4,5,0,7,8,9,8,11,12,8,14,15};
static constexpr int kChi[NBONE] = {1,2,3,4,5,6,7,8,9,10,11,12,13,14,15,16};

// 16-bit A/B operand layout (ISA 7.12.2): element idx of v16h for lane-half `half`
// maps to K = (v<4 ? 2v+j : 16+2(v-4)+j) + 8*half,  v=idx>>1, j=idx&1.
__device__ __forceinline__ int kmap(int idx, int half) {
  int v = idx >> 1, j = idx & 1;
  return (v < 4 ? 2*v + j : 16 + 2*(v - 4) + j) + 8*half;
}

__global__ __launch_bounds__(256)
void bone_gate_wmma_kernel(const float* __restrict__ pred,
                           const float* __restrict__ gbl,
                           const float* __restrict__ anc,
                           const float* __restrict__ W1,
                           const float* __restrict__ W2,
                           float* __restrict__ out,
                           int B)
{
  // Per-wave private LDS slices (8 waves / 256-thread block).
  __shared__ _Float16 s_x[8][32*32];  // x rows (f16) for A-operand gather
  __shared__ _Float16 s_h[8][32*16];  // hidden activations (f16)
  __shared__ float    s_g[8][32*32];  // gates (f32: avoids cvt round-trip)

  const int lane = threadIdx.x & 31;
  const int wid  = threadIdx.x >> 5;
  const int half = lane >> 4;
  const int l16  = lane & 15;

  // Build B-operands once: W1 (32x16), W2 halves (16x32 split into two 16-col
  // tiles, K padded from 16 to 32 with zeros).
  v16h bW1, bW2a, bW2b;
  #pragma unroll
  for (int idx = 0; idx < 16; ++idx) {
    int k = kmap(idx, half);
    bW1[idx]  = (_Float16)W1[k*16 + l16];
    bW2a[idx] = (k < 16) ? (_Float16)W2[k*32 + l16]      : (_Float16)0.0f;
    bW2b[idx] = (k < 16) ? (_Float16)W2[k*32 + 16 + l16] : (_Float16)0.0f;
  }

  _Float16* xb = s_x[wid];
  _Float16* hb = s_h[wid];
  float*    gb = s_g[wid];

  for (int base = blockIdx.x * 256; base < B; base += gridDim.x * 256) {
    const int e      = base + wid*32 + lane;
    const bool valid = (e < B);
    const int eo     = valid ? e : (B - 1);

    // ---------------- Phase 1: geometry -> pre-gate move_bl (x row) --------
    float p[51];
    const float* pj = pred + (long long)eo * 51;
    #pragma unroll
    for (int i = 0; i < 51; ++i) p[i] = pj[i];
    float a[17];
    const float* ap = anc + (long long)eo * 17;
    #pragma unroll
    for (int i = 0; i < 17; ++i) a[i] = ap[i];
    const float* gp = gbl + (long long)eo * 16;

    float u[48], x[32];
    #pragma unroll
    for (int i = 0; i < NBONE; ++i) {
      const int pa = kPar[i], ch = kChi[i];
      float dx = p[pa*3+0] - p[ch*3+0];
      float dy = p[pa*3+1] - p[ch*3+1];
      float dz = p[pa*3+2] - p[ch*3+2];
      float n   = __builtin_sqrtf(dx*dx + dy*dy + dz*dz);
      float inv = 1.0f / (n + FEPS);
      u[3*i+0] = dx*inv; u[3*i+1] = dy*inv; u[3*i+2] = dz*inv;
      float dis = gp[i] - n;
      float rs  = 1.0f / (a[pa] + a[ch] + FEPS);
      x[2*i+0] = dis * a[ch] * rs;
      x[2*i+1] = dis * a[pa] * rs;
    }

    // Mirror own x row to LDS (f16) for the WMMA A-operand gather.
    #pragma unroll
    for (int i = 0; i < 32; ++i) xb[lane*32 + i] = (_Float16)x[i];

    // ---------------- Phase 2: gate MLP on the WMMA units -------------------
    // A-tiles: batch rows 0-15 (a0) and 16-31 (a1) of this wave's chunk.
    v16h a0, a1;
    #pragma unroll
    for (int idx = 0; idx < 16; ++idx) {
      int k = kmap(idx, half);
      a0[idx] = xb[l16*32 + k];
      a1[idx] = xb[(16 + l16)*32 + k];
    }
    v8f z = {};
    v8f h0 = __builtin_amdgcn_wmma_f32_16x16x32_f16(false, a0, false, bW1, (short)0, z, false, false);
    v8f h1 = __builtin_amdgcn_wmma_f32_16x16x32_f16(false, a1, false, bW1, (short)0, z, false, false);

    // ReLU, stage hidden (D layout: lane holds column n=l16, rows r+8*half).
    #pragma unroll
    for (int r = 0; r < 8; ++r) {
      int m = r + 8*half;
      hb[m*16 + l16]        = (_Float16)__builtin_fmaxf(h0[r], 0.0f);
      hb[(16 + m)*16 + l16] = (_Float16)__builtin_fmaxf(h1[r], 0.0f);
    }

    // Re-gather hidden as A-operand (K padded 16->32 with zeros).
    v16h c0, c1;
    #pragma unroll
    for (int idx = 0; idx < 16; ++idx) {
      int k = kmap(idx, half);
      _Float16 hv0 = (_Float16)0.0f, hv1 = (_Float16)0.0f;
      if (k < 16) { hv0 = hb[l16*16 + k]; hv1 = hb[(16 + l16)*16 + k]; }
      c0[idx] = hv0; c1[idx] = hv1;
    }
    v8f g00 = __builtin_amdgcn_wmma_f32_16x16x32_f16(false, c0, false, bW2a, (short)0, z, false, false);
    v8f g01 = __builtin_amdgcn_wmma_f32_16x16x32_f16(false, c0, false, bW2b, (short)0, z, false, false);
    v8f g10 = __builtin_amdgcn_wmma_f32_16x16x32_f16(false, c1, false, bW2a, (short)0, z, false, false);
    v8f g11 = __builtin_amdgcn_wmma_f32_16x16x32_f16(false, c1, false, bW2b, (short)0, z, false, false);

    // Sigmoid; store gates as f32 (no f16 round-trip). Safe reuse ordering:
    // LDS ops are in-order within a wave (DScnt).
    #pragma unroll
    for (int r = 0; r < 8; ++r) {
      int m = r + 8*half;
      gb[m*32 + l16]             = 1.0f / (1.0f + __expf(-g00[r]));
      gb[m*32 + 16 + l16]        = 1.0f / (1.0f + __expf(-g01[r]));
      gb[(16 + m)*32 + l16]      = 1.0f / (1.0f + __expf(-g10[r]));
      gb[(16 + m)*32 + 16 + l16] = 1.0f / (1.0f + __expf(-g11[r]));
    }

    // ---------------- Phase 3: apply gated forces, write output -------------
    #pragma unroll
    for (int i = 0; i < NBONE; ++i) {
      float g0 = gb[lane*32 + 2*i + 0];
      float g1 = gb[lane*32 + 2*i + 1];
      float f0 = x[2*i+0] * g0;
      float f1 = x[2*i+1] * g1;
      const int pa = kPar[i], ch = kChi[i];
      p[pa*3+0] += f0*u[3*i+0]; p[pa*3+1] += f0*u[3*i+1]; p[pa*3+2] += f0*u[3*i+2];
      p[ch*3+0] -= f1*u[3*i+0]; p[ch*3+1] -= f1*u[3*i+1]; p[ch*3+2] -= f1*u[3*i+2];
    }
    if (valid) {
      float* op = out + (long long)e * 51;
      #pragma unroll
      for (int i = 0; i < 51; ++i) op[i] = p[i];
    }
  }
}

extern "C" void kernel_launch(void* const* d_in, const int* in_sizes, int n_in,
                              void* d_out, int out_size, void* d_ws, size_t ws_size,
                              hipStream_t stream) {
  (void)n_in; (void)out_size; (void)d_ws; (void)ws_size;
  const float* pred = (const float*)d_in[0];  // (B,17,3)
  const float* gbl  = (const float*)d_in[1];  // (B,16)
  const float* anc  = (const float*)d_in[2];  // (B,17)
  const float* W1   = (const float*)d_in[3];  // (32,16)
  const float* W2   = (const float*)d_in[4];  // (16,32)
  float* out = (float*)d_out;                 // (B,17,3)

  const int B = in_sizes[0] / 51;
  const int blocks = (B + 255) / 256;         // 256 elems per block-iteration
  bone_gate_wmma_kernel<<<blocks, 256, 0, stream>>>(pred, gbl, anc, W1, W2, out, B);
}
